// TreeLSTMRecNN_3839700762988
// MI455X (gfx1250) — matrix operand order repaired and checked by reading, not compiled, not run on hardware
//
#include <hip/hip_runtime.h>
#include <stdint.h>

// ---- problem constants (from reference) ----
#define B_  16
#define L_  4096
#define D_  256
#define H_  256
#define C_  2

typedef __attribute__((ext_vector_type(16))) __bf16          v16bf;
typedef __attribute__((ext_vector_type(8)))  float           v8f;
typedef __attribute__((ext_vector_type(8)))  unsigned short  v8u16;

__device__ __forceinline__ unsigned short f2bf(float f){
  unsigned int u = __float_as_uint(f);
  u += 0x7FFFu + ((u >> 16) & 1u);         // round-to-nearest-even
  return (unsigned short)(u >> 16);
}
__device__ __forceinline__ float bf2f(unsigned short h){
  return __uint_as_float(((unsigned int)h) << 16);
}
__device__ __forceinline__ float sigm(float x){ return 1.0f / (1.0f + __expf(-x)); }

union FragU { v8u16 h[2]; v16bf v; };

// A fragment (16x32 bf16) per ISA layout: lane L holds row M=L&15;
// lanes 0-15: K = kt*32 + {0..7, 16..23}; lanes 16-31: +8.
// Caller passes pointer already offset to sA[M*stride + kt*32 + kb].
__device__ __forceinline__ v16bf afrag(const unsigned short* p){
  FragU u;
  u.h[0] = *(const v8u16*)(p);
  u.h[1] = *(const v8u16*)(p + 16);
  return u.v;
}

// B fragment (32x16 bf16) from pre-packed weights: per-lane contiguous 16 elems.
__device__ __forceinline__ v16bf bfrag(const unsigned short* __restrict__ Wp,
                                       int nt, int kt, int KT, int lane){
  const unsigned short* p = Wp + ((((size_t)nt * KT + kt) * 32) + lane) * 16;
  FragU u;
  u.h[0] = ((const v8u16*)p)[0];
  u.h[1] = ((const v8u16*)p)[1];
  return u.v;
}

// ---- weight repack: W_proj (256x256 row-major f32) -> B-fragment order bf16 ----
__global__ void pack_wproj(const float* __restrict__ W, unsigned short* __restrict__ out){
  int tid  = blockIdx.x * 256 + threadIdx.x;     // 65536
  int e    = tid & 15;
  int lane = (tid >> 4) & 31;
  int kt   = (tid >> 9) & 7;                     // KT = 8
  int nt   = tid >> 12;                          // NT = 16
  int K = kt * 32 + ((lane & 16) ? 8 : 0) + (e & 7) + ((e & 8) ? 16 : 0);
  int n = nt * 16 + (lane & 15);
  out[tid] = f2bf(W[K * H_ + n]);
}

// ---- fused gate weights: 5 x (512x256) -> (512 x 1280) packed bf16 ----
// gate order: 0=fl 1=fr 2=i 3=o 4=c
__global__ void pack_wall(const float* __restrict__ Wfl, const float* __restrict__ Wfr,
                          const float* __restrict__ Wi,  const float* __restrict__ Wo,
                          const float* __restrict__ Wc,  unsigned short* __restrict__ out){
  int tid  = blockIdx.x * 256 + threadIdx.x;     // 655360
  int e    = tid & 15;
  int lane = (tid >> 4) & 31;
  int kt   = (tid >> 9) & 15;                    // KT = 16
  int nt   = tid >> 13;                          // NT = 80
  int col  = nt * 16 + (lane & 15);
  int g    = col >> 8;
  int n    = col & 255;
  int K    = kt * 32 + ((lane & 16) ? 8 : 0) + (e & 7) + ((e & 8) ? 16 : 0);
  const float* W = (g == 0) ? Wfl : (g == 1) ? Wfr : (g == 2) ? Wi : (g == 3) ? Wo : Wc;
  out[tid] = f2bf(W[K * H_ + n]);
}

// ---- leaf kernel: gather + projection GEMM + tanh ----
// 16 rows/block, 8 waves, 2 N-tiles per wave (N = 256).
__global__ void __launch_bounds__(256)
leaf_proj(const int* __restrict__ ids, const float* __restrict__ emb,
          const unsigned short* __restrict__ WpPack, const float* __restrict__ bias,
          unsigned short* __restrict__ h_out, float* __restrict__ c_out){
  __shared__ __align__(16) unsigned short sA[16 * 256];
  int t  = threadIdx.x;
  int r0 = blockIdx.x * 16;
  { // stage A (bf16) into LDS: 16 rows x 256, thread copies 16 elems
    int rl = t >> 4, seg = t & 15;
    int id = ids[r0 + rl];
    const float* src = emb + (size_t)id * D_ + seg * 16;
    unsigned short* dst = sA + rl * 256 + seg * 16;
    #pragma unroll
    for (int i = 0; i < 16; ++i) dst[i] = f2bf(src[i]);
  }
  __syncthreads();
  int lane = t & 31, wave = t >> 5;
  v8f acc[2] = {};
  const unsigned short* arow = sA + (lane & 15) * 256 + ((lane & 16) ? 8 : 0);
  #pragma unroll
  for (int kt = 0; kt < 8; ++kt){
    v16bf a = afrag(arow + kt * 32);
    #pragma unroll
    for (int tc = 0; tc < 2; ++tc){
      int nt = wave * 2 + tc;
      v16bf b = bfrag(WpPack, nt, kt, 8, lane);
      acc[tc] = __builtin_amdgcn_wmma_f32_16x16x32_bf16(
          false, a, false, b, (short)0, acc[tc], false, false);
    }
  }
  #pragma unroll
  for (int tc = 0; tc < 2; ++tc){
    int nt = wave * 2 + tc;
    int n  = nt * 16 + (lane & 15);
    float bp = bias[n];
    #pragma unroll
    for (int j = 0; j < 8; ++j){
      int rl = j + ((lane & 16) ? 8 : 0);
      size_t idx = (size_t)(r0 + rl) * H_ + n;
      float h = acc[tc][j] + bp;
      h_out[idx] = f2bf(h);
      c_out[idx] = tanhf(h);
    }
  }
}

// ---- tree level: 5-gate fused GEMM (rows x 512 @ 512 x 1280) + LSTM combine ----
// Wave w owns output columns m in {w, w+8} across all 5 gates -> epilogue in regs.
__global__ void __launch_bounds__(256)
tree_level(const unsigned short* __restrict__ h_in, const float* __restrict__ c_in,
           unsigned short* __restrict__ h_out, float* __restrict__ c_out,
           const unsigned short* __restrict__ Wp,
           const float* __restrict__ bfl_, const float* __restrict__ bfr_,
           const float* __restrict__ bi_,  const float* __restrict__ bo_,
           const float* __restrict__ bc_,
           int N, int halfN){
  __shared__ __align__(16) unsigned short sA[16 * 512];
  int t  = threadIdx.x;
  int r0 = blockIdx.x * 16;
  { // stage comb = [h_left | h_right] rows: 16 x 512 bf16; thread copies 64 B
    int rl = t >> 4, seg = t & 15;
    int R = r0 + rl;
    int b = R / halfN;
    int k = R - b * halfN;
    size_t srcrow = (size_t)(b * N + 2 * k + (seg >> 3)) * H_;
    const uint4* s4 = (const uint4*)(h_in + srcrow + (size_t)(seg & 7) * 32);
    uint4* d4 = (uint4*)(sA + rl * 512 + seg * 32);
    d4[0] = s4[0]; d4[1] = s4[1]; d4[2] = s4[2]; d4[3] = s4[3];
  }
  __syncthreads();
  int lane = t & 31, wave = t >> 5;
  v8f acc[5][2] = {};
  const unsigned short* arow = sA + (lane & 15) * 512 + ((lane & 16) ? 8 : 0);
  for (int kt = 0; kt < 16; ++kt){
    v16bf a = afrag(arow + kt * 32);
    #pragma unroll
    for (int g = 0; g < 5; ++g){
      #pragma unroll
      for (int tc = 0; tc < 2; ++tc){
        int nt = g * 16 + wave + 8 * tc;
        v16bf b = bfrag(Wp, nt, kt, 16, lane);
        acc[g][tc] = __builtin_amdgcn_wmma_f32_16x16x32_bf16(
            false, a, false, b, (short)0, acc[g][tc], false, false);
      }
    }
  }
  #pragma unroll
  for (int tc = 0; tc < 2; ++tc){
    int m = wave + 8 * tc;
    int n = m * 16 + (lane & 15);
    float vfl = bfl_[n], vfr = bfr_[n], vi = bi_[n], vo = bo_[n], vc = bc_[n];
    #pragma unroll
    for (int j = 0; j < 8; ++j){
      int rl = j + ((lane & 16) ? 8 : 0);
      int R = r0 + rl;
      int b = R / halfN;
      int k = R - b * halfN;
      size_t il = (size_t)(b * N + 2 * k) * H_ + n;
      float cl = c_in[il], cr = c_in[il + H_];
      float fl = sigm(acc[0][tc][j] + vfl);
      float fr = sigm(acc[1][tc][j] + vfr);
      float ig = sigm(acc[2][tc][j] + vi);
      float og = sigm(acc[3][tc][j] + vo);
      float cg = tanhf(acc[4][tc][j] + vc);
      float cn = fl * cl + fr * cr + ig * cg;
      size_t io = (size_t)R * H_ + n;
      c_out[io] = cn;
      h_out[io] = f2bf(og * tanhf(cn));
    }
  }
}

// ---- classifier head: (16 x 256) @ (256 x 2) + b ----
__global__ void cls_head(const unsigned short* __restrict__ hroot,
                         const float* __restrict__ Wcls, const float* __restrict__ bcls,
                         float* __restrict__ out){
  int t = threadIdx.x;
  if (t >= B_ * C_) return;
  int b = t >> 1, c = t & 1;
  float s = bcls[c];
  #pragma unroll 4
  for (int d = 0; d < H_; ++d) s += bf2f(hroot[b * H_ + d]) * Wcls[d * C_ + c];
  out[t] = s;
}

extern "C" void kernel_launch(void* const* d_in, const int* in_sizes, int n_in,
                              void* d_out, int out_size, void* d_ws, size_t ws_size,
                              hipStream_t stream){
  const int*   ids   = (const int*)  d_in[0];
  const float* emb   = (const float*)d_in[1];
  const float* Wproj = (const float*)d_in[2];
  const float* bproj = (const float*)d_in[3];
  const float* Wfl   = (const float*)d_in[4];
  const float* bfl   = (const float*)d_in[5];
  const float* Wfr   = (const float*)d_in[6];
  const float* bfr   = (const float*)d_in[7];
  const float* Wi    = (const float*)d_in[8];
  const float* bi    = (const float*)d_in[9];
  const float* Wo    = (const float*)d_in[10];
  const float* bo    = (const float*)d_in[11];
  const float* Wc    = (const float*)d_in[12];
  const float* bc    = (const float*)d_in[13];
  const float* Wcls  = (const float*)d_in[14];
  const float* bcls  = (const float*)d_in[15];
  float* out = (float*)d_out;

  char* ws = (char*)d_ws;
  size_t off = 0;
  auto take = [&](size_t bytes) -> char* {
    char* p = ws + off; off += (bytes + 255) & ~(size_t)255; return p;
  };
  unsigned short* WpPack   = (unsigned short*)take((size_t)D_ * H_ * 2);          // 128 KB
  unsigned short* WallPack = (unsigned short*)take((size_t)2 * H_ * 5 * H_ * 2);  // 1.25 MB
  unsigned short* hA = (unsigned short*)take((size_t)B_ * L_ * H_ * 2);           // 32 MB
  unsigned short* hB = (unsigned short*)take((size_t)B_ * (L_ / 2) * H_ * 2);     // 16 MB
  float* cA = (float*)take((size_t)B_ * L_ * H_ * 4);                             // 64 MB
  float* cB = (float*)take((size_t)B_ * (L_ / 2) * H_ * 4);                       // 32 MB
  (void)ws_size; (void)in_sizes; (void)n_in; (void)out_size;

  pack_wproj<<<(D_ * H_) / 256, 256, 0, stream>>>(Wproj, WpPack);
  pack_wall<<<(2 * H_ * 5 * H_) / 256, 256, 0, stream>>>(Wfl, Wfr, Wi, Wo, Wc, WallPack);
  leaf_proj<<<(B_ * L_) / 16, 256, 0, stream>>>(ids, emb, WpPack, bproj, hA, cA);

  unsigned short *hin = hA, *hout = hB;
  float *cin = cA, *cout = cB;
  int N = L_;
  while (N > 1){
    int halfN = N >> 1;
    int rows  = B_ * halfN;                     // always a multiple of 16
    tree_level<<<rows / 16, 256, 0, stream>>>(hin, cin, hout, cout, WallPack,
                                              bfl, bfr, bi, bo, bc, N, halfN);
    unsigned short* th = hin; hin = hout; hout = th;
    float* tf = cin; cin = cout; cout = tf;
    N = halfN;
  }
  cls_head<<<1, 32, 0, stream>>>(hin, Wcls, bcls, out);
}
